// LoopCoderAttention_27565100106317
// MI455X (gfx1250) — matrix operand order repaired and checked by reading, not compiled, not run on hardware
//
#include <hip/hip_runtime.h>
#include <hip/hip_bf16.h>

#define T_DIM 2048
#define HID   4096
#define NH    32
#define NKV   8
#define HD    128
#define WIN   64
#define QKV_N 6144                       // (32 + 2*8) * 128
#define SCALE 0.08838834764831845f       // 128^-0.5
#define NEGV  -1000000000.0f

typedef __attribute__((ext_vector_type(16))) __bf16 v16bf;
typedef __attribute__((ext_vector_type(8)))  float  v8f;

struct bpair_t { uint4 a, b; };

__device__ inline v16bf make_frag(uint4 x, uint4 y) {
  bpair_t t{x, y};
  return __builtin_bit_cast(v16bf, t);
}
__device__ inline __bf16 f2bf(float f) {
  __hip_bfloat16 h = __float2bfloat16(f);
  return __builtin_bit_cast(__bf16, h);
}
__device__ inline float bf2f(__bf16 b) {
  __hip_bfloat16 h = __builtin_bit_cast(__hip_bfloat16, b);
  return __bfloat162float(h);
}

// ---------------------------------------------------------------- convert
__global__ void k_f32_to_bf16(const float* __restrict__ src,
                              __bf16* __restrict__ dst, long n) {
  long i = (long)blockIdx.x * blockDim.x + threadIdx.x;
  long stride = (long)gridDim.x * blockDim.x;
  for (; i < n; i += stride) dst[i] = f2bf(src[i]);
}

// ---------------------------------------------------------------- V transpose
// v (T, NKV*HD) f32  ->  vt (NKV*HD, T) bf16
__global__ void k_transpose_v(const float* __restrict__ v, __bf16* __restrict__ vt) {
  int col = blockIdx.x;  // g*HD + d, 1024 total
  for (int t = threadIdx.x; t < T_DIM; t += blockDim.x)
    vt[(size_t)col * T_DIM + t] = f2bf(v[(size_t)t * (NKV * HD) + col]);
}

// ---------------------------------------------------------------- GEMM (NT)
// C[M,N] = A[M,K](bf16 rm) * B[N,K](bf16 rm)^T, f32 out.
// Block: 256 thr = 8 waves (4m x 2n); wave tile 32x64 (2x4 WMMA), block 128x128.
// K loop unrolled x2 with statically-named double buffers (no v_movrels).
// Requires K % 64 == 0.
__global__ __launch_bounds__(256)
void k_gemm_bf16_nt(const __bf16* __restrict__ A, const __bf16* __restrict__ B,
                    float* __restrict__ C, int M, int N, int K) {
  const int lane = threadIdx.x & 31;
  const int w    = threadIdx.x >> 5;
  const int ln   = lane & 15, hi = lane >> 4;
  const int m_base = blockIdx.x * 128 + (w & 3) * 32;
  const int n_base = blockIdx.y * 128 + (w >> 2) * 64;

  const __bf16* ap[2];
  const __bf16* bp[4];
#pragma unroll
  for (int mt = 0; mt < 2; ++mt) ap[mt] = A + (size_t)(m_base + mt * 16 + ln) * K;
#pragma unroll
  for (int nt = 0; nt < 4; ++nt) bp[nt] = B + (size_t)(n_base + nt * 16 + ln) * K;

  v16bf af0[2], bf0[4], af1[2], bf1[4];
  v8f acc[2][4] = {};

#define LOAD_TILES(AF, BF, K0)                                                  \
  {                                                                             \
    _Pragma("unroll") for (int mt = 0; mt < 2; ++mt) {                          \
      const __bf16* pA = ap[mt] + (K0);                                         \
      uint4 x = *reinterpret_cast<const uint4*>(pA + hi * 16);                  \
      uint4 y = *reinterpret_cast<const uint4*>(pA + hi * 16 + 8);              \
      AF[mt] = make_frag(x, y);                                                 \
      __builtin_prefetch(pA + 64, 0, 3);                                        \
    }                                                                           \
    _Pragma("unroll") for (int nt = 0; nt < 4; ++nt) {                          \
      const __bf16* pB = bp[nt] + (K0) + hi * 16;                               \
      uint4 x = *reinterpret_cast<const uint4*>(pB);                            \
      uint4 y = *reinterpret_cast<const uint4*>(pB + 8);                        \
      BF[nt] = make_frag(x, y);                                                 \
      __builtin_prefetch(pB + 64, 0, 3);                                        \
    }                                                                           \
  }

#define COMPUTE(AF, BF)                                                         \
  {                                                                             \
    _Pragma("unroll") for (int mt = 0; mt < 2; ++mt)                            \
      _Pragma("unroll") for (int nt = 0; nt < 4; ++nt)                          \
        acc[mt][nt] = __builtin_amdgcn_wmma_f32_16x16x32_bf16(                  \
            false, AF[mt], false, BF[nt], (short)0, acc[mt][nt], false, false); \
  }

  LOAD_TILES(af0, bf0, 0)
  for (int k0 = 0; k0 < K; k0 += 64) {
    LOAD_TILES(af1, bf1, k0 + 32)
    COMPUTE(af0, bf0)
    if (k0 + 64 < K) LOAD_TILES(af0, bf0, k0 + 64)
    COMPUTE(af1, bf1)
  }
#undef LOAD_TILES
#undef COMPUTE

#pragma unroll
  for (int mt = 0; mt < 2; ++mt)
#pragma unroll
    for (int r = 0; r < 8; ++r) {
      const int row = m_base + mt * 16 + r + 8 * hi;
      float* cp = C + (size_t)row * N + n_base + ln;
#pragma unroll
      for (int nt = 0; nt < 4; ++nt) cp[nt * 16] = acc[mt][nt][r];
    }
}

// ---------------------------------------------------------------- RoPE + split
// Writes q (T,H*D) bf16, k (T,NKV*HD) bf16, and local V transposed (NKV*HD, T).
__global__ void k_rope_split(const float* __restrict__ qkv,
                             const int* __restrict__ pos,
                             __bf16* __restrict__ q, __bf16* __restrict__ k,
                             __bf16* __restrict__ vt) {
  const int t = blockIdx.x;
  const float p = (float)pos[t];
  const float* row = qkv + (size_t)t * QKV_N;
  for (int idx = threadIdx.x; idx < (NH + NKV) * 64; idx += blockDim.x) {
    int h = idx >> 6, j = idx & 63;
    float inv = __powf(10000.0f, -(float)j * (1.0f / 64.0f));
    float f = p * inv;
    float c = __cosf(f), s = __sinf(f);
    int base = h * HD;  // q heads then k heads contiguous in qkv row
    float x1 = row[base + j], x2 = row[base + 64 + j];
    float o1 = x1 * c - x2 * s, o2 = x2 * c + x1 * s;
    if (h < NH) {
      q[(size_t)t * HID + h * HD + j]      = f2bf(o1);
      q[(size_t)t * HID + h * HD + 64 + j] = f2bf(o2);
    } else {
      int hk = h - NH;
      k[(size_t)t * (NKV * HD) + hk * HD + j]      = f2bf(o1);
      k[(size_t)t * (NKV * HD) + hk * HD + 64 + j] = f2bf(o2);
    }
  }
  for (int idx = threadIdx.x; idx < NKV * HD; idx += blockDim.x)
    vt[(size_t)idx * T_DIM + t] = f2bf(row[(NH + NKV) * HD + idx]);
}

// ---------------------------------------------------------------- gate
__global__ void k_gate(const __bf16* __restrict__ q, const float* __restrict__ w_gate,
                       const float* __restrict__ b_gate, float* __restrict__ gate) {
  int i = blockIdx.x * blockDim.x + threadIdx.x;  // i = t*NH + h
  if (i >= T_DIM * NH) return;
  int t = i >> 5, h = i & 31;
  const __bf16* qp = q + (size_t)t * HID + h * HD;
  const float*  wp = w_gate + h * HD;
  float acc = 0.f;
#pragma unroll 4
  for (int d = 0; d < HD; ++d) acc += bf2f(qp[d]) * wp[d];
  acc += b_gate[h];
  gate[i] = 1.0f / (1.0f + __expf(-acc));
}

// ---------------------------------------------------------------- flash attention
// One wave = one 16-row Q block for head hh. Km row-major (T, NKV*HD);
// Vt transposed (NKV*HD, T). local_mode=0: causal; 1: causal window WIN.
__global__ __launch_bounds__(128)
void k_flash(const __bf16* __restrict__ Q, const __bf16* __restrict__ Km,
             const __bf16* __restrict__ Vt, float* __restrict__ Out,
             int local_mode) {
  __shared__ __bf16 plds[4][16][32];  // per-wave P tile (16 rows x 32 keys)
  const int lane = threadIdx.x & 31;
  const int w    = threadIdx.x >> 5;
  const int ln   = lane & 15, hi = lane >> 4;
  const int hh = blockIdx.y, g = hh >> 2;
  const int q0 = (blockIdx.x * 4 + w) * 16;

  // Q fragments: A rows m = q0+ln, contraction = head dim (4 chunks of 32)
  v16bf qf[4];
#pragma unroll
  for (int dk = 0; dk < 4; ++dk) {
    const __bf16* p = Q + (size_t)(q0 + ln) * HID + hh * HD + dk * 32;
    uint4 x = *reinterpret_cast<const uint4*>(p + hi * 16);
    uint4 y = *reinterpret_cast<const uint4*>(p + hi * 16 + 8);
    qf[dk] = make_frag(x, y);
  }

  float rmax[8], rsum[8];
  v8f acc[8] = {};
#pragma unroll
  for (int r = 0; r < 8; ++r) { rmax[r] = -3.0e38f; rsum[r] = 0.f; }

  int kb_begin = 0;
  if (local_mode) {
    int s = q0 - (WIN - 1);
    if (s < 0) s = 0;
    kb_begin = s & ~31;
  }
  const int kb_last = (q0 + 15) & ~31;

  for (int kb = kb_begin; kb <= kb_last; kb += 32) {
    // ---- S = Q * K^T for 32 keys (two 16x16 tiles)
    v8f s[2] = {};
#pragma unroll
    for (int st = 0; st < 2; ++st) {
      int key = kb + st * 16 + ln;  // B column n -> key position
#pragma unroll
      for (int dk = 0; dk < 4; ++dk) {
        const __bf16* p = Km + (size_t)key * (NKV * HD) + g * HD + dk * 32 + hi * 16;
        uint4 x = *reinterpret_cast<const uint4*>(p);
        uint4 y = *reinterpret_cast<const uint4*>(p + 8);
        v16bf kf = make_frag(x, y);
        s[st] = __builtin_amdgcn_wmma_f32_16x16x32_bf16(
            false, qf[dk], false, kf, (short)0, s[st], false, false);
      }
    }
    // ---- scale + mask (row = q0 + r + 8*hi, key = kb + st*16 + ln)
#pragma unroll
    for (int st = 0; st < 2; ++st) {
      int key = kb + st * 16 + ln;
#pragma unroll
      for (int r = 0; r < 8; ++r) {
        int mrow = q0 + r + 8 * hi;
        bool ok = (key <= mrow) && (!local_mode || key >= mrow - (WIN - 1));
        s[st][r] = ok ? s[st][r] * SCALE : NEGV;
      }
    }
    // ---- online softmax; rows span 16-lane halves
#pragma unroll
    for (int r = 0; r < 8; ++r) {
      float mx = fmaxf(s[0][r], s[1][r]);
#pragma unroll
      for (int off = 1; off < 16; off <<= 1)
        mx = fmaxf(mx, __shfl_xor(mx, off, 32));
      float nm = fmaxf(rmax[r], mx);
      float p0 = __expf(s[0][r] - nm);
      float p1 = __expf(s[1][r] - nm);
      float sm = p0 + p1;
#pragma unroll
      for (int off = 1; off < 16; off <<= 1)
        sm += __shfl_xor(sm, off, 32);
      float alpha = __expf(rmax[r] - nm);
      rsum[r] = rsum[r] * alpha + sm;
      rmax[r] = nm;
#pragma unroll
      for (int nt = 0; nt < 8; ++nt) acc[nt][r] = acc[nt][r] * alpha;
      plds[w][r + 8 * hi][ln]      = f2bf(p0);
      plds[w][r + 8 * hi][16 + ln] = f2bf(p1);
    }
    // ---- re-layout P as A fragment via LDS (rows 64B-aligned)
    const __bf16* prow = &plds[w][ln][0];
    uint4 px = *reinterpret_cast<const uint4*>(prow + hi * 16);
    uint4 py = *reinterpret_cast<const uint4*>(prow + hi * 16 + 8);
    v16bf pf = make_frag(px, py);
    // ---- O += P * V : 8 d-chunks; Vt gives contiguous key runs per d row
#pragma unroll
    for (int nt = 0; nt < 8; ++nt) {
      const __bf16* vp = Vt + (size_t)(g * HD + nt * 16 + ln) * T_DIM + kb + hi * 16;
      uint4 x = *reinterpret_cast<const uint4*>(vp);
      uint4 y = *reinterpret_cast<const uint4*>(vp + 8);
      v16bf vf = make_frag(x, y);
      acc[nt] = __builtin_amdgcn_wmma_f32_16x16x32_bf16(
          false, pf, false, vf, (short)0, acc[nt], false, false);
    }
  }
  // ---- epilogue: normalize + store (one base pointer per row, imm offsets)
#pragma unroll
  for (int r = 0; r < 8; ++r) {
    const int row = q0 + r + 8 * hi;
    float* op = Out + (size_t)row * HID + hh * HD + ln;
    float inv = 1.0f / rsum[r];
#pragma unroll
    for (int nt = 0; nt < 8; ++nt) op[nt * 16] = acc[nt][r] * inv;
  }
}

// ---------------------------------------------------------------- combine
__global__ void k_combine(const float* __restrict__ outg,
                          const float* __restrict__ outl,
                          const float* __restrict__ gate,
                          __bf16* __restrict__ comb) {
  size_t i = (size_t)blockIdx.x * blockDim.x + threadIdx.x;
  if (i >= (size_t)T_DIM * HID) return;
  int t = (int)(i >> 12);
  int h = (int)((i & 4095) >> 7);
  float gv = gate[t * NH + h];
  comb[i] = f2bf(gv * outg[i] + (1.0f - gv) * outl[i]);
}

// ---------------------------------------------------------------- launch
extern "C" void kernel_launch(void* const* d_in, const int* in_sizes, int n_in,
                              void* d_out, int out_size, void* d_ws, size_t ws_size,
                              hipStream_t stream) {
  (void)in_sizes; (void)n_in; (void)out_size; (void)ws_size;
  const int*   positions = (const int*)d_in[0];
  const float* hidden    = (const float*)d_in[1];
  const float* k_global  = (const float*)d_in[2];
  const float* v_global  = (const float*)d_in[3];
  const float* w_qkv     = (const float*)d_in[4];
  const float* w_o       = (const float*)d_in[5];
  const float* w_gate    = (const float*)d_in[6];
  const float* b_gate    = (const float*)d_in[7];
  float* out = (float*)d_out;

  char* ws = (char*)d_ws;
  const size_t MB = 1ull << 20;
  __bf16* hs_bf   = (__bf16*)(ws + 0);        // 16 MB : hidden bf16
  __bf16* wqkv_bf = (__bf16*)(ws + 16 * MB);  // 48 MB
  __bf16* wo_bf   = (__bf16*)(ws + 64 * MB);  // 32 MB
  __bf16* kg_bf   = (__bf16*)(ws + 96 * MB);  //  4 MB : k_global bf16 (row-major)
  __bf16* vgT_bf  = (__bf16*)(ws + 100 * MB); //  4 MB : v_global^T bf16
  float*  qkv     = (float*) (ws + 104 * MB); // 48 MB
  __bf16* q_bf    = (__bf16*)(ws + 152 * MB); // 16 MB
  __bf16* k_bf    = (__bf16*)(ws + 168 * MB); //  4 MB : local k (row-major)
  __bf16* vT_bf   = (__bf16*)(ws + 172 * MB); //  4 MB : local v^T
  float*  gate    = (float*) (ws + 176 * MB); // 256 KB
  float*  outg    = (float*) (ws + 177 * MB); // 32 MB
  float*  outl    = (float*) (ws + 209 * MB); // 32 MB
  __bf16* comb    = (__bf16*)(ws + 241 * MB); // 16 MB

  auto cvt = [&](const float* s, __bf16* d, long n) {
    long b = (n + 1023) / 1024;
    int blocks = (int)(b > 16384 ? 16384 : b);
    k_f32_to_bf16<<<blocks, 256, 0, stream>>>(s, d, n);
  };
  cvt(hidden,   hs_bf,   (long)T_DIM * HID);
  cvt(w_qkv,    wqkv_bf, (long)QKV_N * HID);
  cvt(w_o,      wo_bf,   (long)HID * HID);
  cvt(k_global, kg_bf,   (long)T_DIM * NKV * HD);
  k_transpose_v<<<NKV * HD, 256, 0, stream>>>(v_global, vgT_bf);

  // qkv = hidden @ w_qkv^T
  k_gemm_bf16_nt<<<dim3(T_DIM / 128, QKV_N / 128), 256, 0, stream>>>(
      hs_bf, wqkv_bf, qkv, T_DIM, QKV_N, HID);

  k_rope_split<<<T_DIM, 256, 0, stream>>>(qkv, positions, q_bf, k_bf, vT_bf);
  k_gate<<<(T_DIM * NH) / 256, 256, 0, stream>>>(q_bf, w_gate, b_gate, gate);

  // global causal attention vs k_global / v_global^T
  k_flash<<<dim3(T_DIM / 64, NH), 128, 0, stream>>>(q_bf, kg_bf, vgT_bf, outg, 0);
  // local sliding-window attention vs RoPE'd k / v^T
  k_flash<<<dim3(T_DIM / 64, NH), 128, 0, stream>>>(q_bf, k_bf, vT_bf, outl, 1);

  k_combine<<<(T_DIM * HID) / 256, 256, 0, stream>>>(outg, outl, gate, comb);

  // out = comb @ w_o^T
  k_gemm_bf16_nt<<<dim3(T_DIM / 128, HID / 128), 256, 0, stream>>>(
      comb, wo_bf, out, T_DIM, HID, HID);
}
